// NAIS_regionEmbedding_77395310674422
// MI455X (gfx1250) — matrix operand order repaired and checked by reading, not compiled, not run on hardware
//
#include <hip/hip_runtime.h>
#include <hip/hip_bf16.h>
#include <math.h>

// CDNA5 wave32 WMMA types
typedef __attribute__((ext_vector_type(16))) _Float16 v16h;
typedef __attribute__((ext_vector_type(8)))  float    v8f;

#define HLEN   200
#define DIMV   128
#define HALFV  64
#define HIDV   64
#define NWAVE  2                       // waves per block (64 threads)
#define NTILE  ((HLEN + 15) / 16)      // 13
#define ASCALE 4096.0f                 // 2^12: lift ~1e-4 products into f16 normal range
#define AINV   (1.0f / ASCALE)

union PackU { unsigned u; _Float16 h[2]; };

__global__ __launch_bounds__(64) void nais_region_fused(
    const int*   __restrict__ history,      // [B,H]
    const int*   __restrict__ target,       // [B]
    const int*   __restrict__ hist_region,  // [B,H]
    const int*   __restrict__ tgt_region,   // [B]
    const float* __restrict__ W_hist,       // [ITEM,64]
    const float* __restrict__ W_tgt,        // [ITEM,64]
    const float* __restrict__ W_reg,        // [REGION,64]
    const float* __restrict__ W1,           // [128,64] row-major
    const float* __restrict__ b1,           // [64]
    const float* __restrict__ W2,           // [64]
    float*       __restrict__ out)          // [B]
{
    // ---- LDS ----
    __shared__ __align__(16) float    tsc[DIMV];                  // target embedding * ASCALE
    __shared__ float                  b1s[HIDV];
    __shared__ float                  w2s[HIDV];
    // W1 pre-swizzled into WMMA B-fragment layout (ISA 05_wmma 32x16 B):
    // frag f = c*4+n; [f][lane][r] = pack(W1[k][col], W1[k+1][col]),
// col = n*16 + (lane&15), k = c*32 + (r<4?0:16) + (lane>>4)*8 + 2*(r&3)
    __shared__ __align__(32) unsigned w1f[16 * 32 * 8];           // 16 KB
    __shared__ __align__(16) float    hbuf[NWAVE][2][16][DIMV];   // double-buffered, 32 KB
    __shared__ __align__(16) float    dummy[DIMV];                // async sink for padded rows
    __shared__ float                  s_exp, s_es;

    const int b    = blockIdx.x;
    const int tid  = threadIdx.x;
    const int lane = tid & 31;
    const int wv   = tid >> 5;
    const int hi   = lane >> 4;     // which 16-lane half
    const int cl   = lane & 15;

    const int tgt = target[b];

    // ---- block prologue ----
    if (tid < HIDV) {
        b1s[tid] = b1[tid];
        w2s[tid] = W2[tid];
    }
    {
        const int tr = tgt_region[b];
        for (int i = tid; i < DIMV; i += 64) {
            const float tv = (i < HALFV) ? W_tgt[(size_t)tgt * HALFV + i]
                                         : W_reg[(size_t)tr  * HALFV + (i - HALFV)];
            tsc[i] = tv * ASCALE;
        }
    }
    for (int e = tid; e < 16 * 32 * 8; e += 64) {
        const int f  = e >> 8;
        const int ln = (e >> 3) & 31;
        const int r  = e & 7;
        const int c  = f >> 2, n = f & 3;
        const int col = n * 16 + (ln & 15);
        const int k   = c * 32 + ((r < 4) ? 0 : 16) + (ln >> 4) * 8 + 2 * (r & 3);
        PackU p;
        p.h[0] = (_Float16)W1[(size_t)k       * HIDV + col];
        p.h[1] = (_Float16)W1[(size_t)(k + 1) * HIDV + col];
        w1f[e] = p.u;
    }
    if (tid == 0) { s_exp = 0.0f; s_es = 0.0f; }
    __syncthreads();

    const int* histRow = history     + (size_t)b * HLEN;
    const int* hregRow = hist_region + (size_t)b * HLEN;

    // Issue 16 async row-gathers (global -> LDS, no VGPR round trip) for one tile.
    // Each instruction moves a full 512B embedding row: lanes 0-15 the W_hist half,
    // lanes 16-31 the W_reg half, 16B per lane. Padded rows go to a dummy sink so
    // the ASYNCcnt wait immediate stays constant; their slots are zeroed via DS.
    auto issue_tile = [&](int tile, float* dstbase) {
        const int kbase = hi * HALFV + cl * 4;
        #pragma unroll 4
        for (int jj = 0; jj < 16; ++jj) {
            const int j = tile * 16 + jj;
            const bool valid = (j < HLEN);
            const float* gp;
            float*       lp;
            if (valid) {
                gp = hi ? (W_reg  + (size_t)hregRow[j] * HALFV + cl * 4)
                        : (W_hist + (size_t)histRow[j] * HALFV + cl * 4);
                lp = dstbase + jj * DIMV + kbase;
            } else {
                gp = W_hist + cl * 4;        // harmless valid address
                lp = &dummy[kbase];          // sink
            }
            const unsigned long long ga = (unsigned long long)(size_t)gp;
            const unsigned           la = (unsigned)(size_t)lp;  // low 32b = LDS offset
            asm volatile("global_load_async_to_lds_b128 %0, %1, off"
                         :: "v"(la), "v"(ga) : "memory");
            if (!valid)
                *(float4*)(dstbase + jj * DIMV + kbase) = make_float4(0.f, 0.f, 0.f, 0.f);
        }
    };

    if (wv < NTILE) issue_tile(wv, &hbuf[wv][0][0][0]);
    int buf = 0;

    for (int tile = wv; tile < NTILE; tile += NWAVE) {
        // prefetch next tile into the other buffer, then drain current tile's asyncs
        const int nt = tile + NWAVE;
        if (nt < NTILE) {
            issue_tile(nt, &hbuf[wv][buf ^ 1][0][0]);
            asm volatile("s_wait_asynccnt 0x10" ::: "memory");  // <=16: current buffer done
        } else {
            asm volatile("s_wait_asynccnt 0x0" ::: "memory");
        }

        v8f acc[4];
        #pragma unroll
        for (int n = 0; n < 4; ++n) acc[n] = (v8f){};
        float sp = 0.0f;                          // scaled partial row-sum of inp
        const float* myrow = &hbuf[wv][buf][cl][0];  // A row M = lane&15 (ISA A layout)

        #pragma unroll
        for (int c = 0; c < 4; ++c) {
            // A fragment, 16-bit 16x32 layout: VGPR r holds K = c*32 + (r<4?0:16) + hi*8 + 2*(r&3)
            v16h a;
            #pragma unroll
            for (int r = 0; r < 8; ++r) {
                const int k = c * 32 + ((r < 4) ? 0 : 16) + hi * 8 + (r & 3) * 2;
                const float2 hv = *(const float2*)&myrow[k];
                const float2 tv = *(const float2*)&tsc[k];   // broadcast within half-wave
                const float px = hv.x * tv.x, py = hv.y * tv.y;
                sp += px + py;
                a[2 * r]     = (_Float16)px;
                a[2 * r + 1] = (_Float16)py;
            }
            #pragma unroll
            for (int n = 0; n < 4; ++n) {
                const v16h bf = *(const v16h*)&w1f[(c * 4 + n) * 256 + lane * 8];
                acc[n] = __builtin_amdgcn_wmma_f32_16x16x32_f16(
                    false, a, false, bf, (short)0, acc[n], false, false);
            }
        }
        buf ^= 1;

        // full row-sum (descaled): lane pair (L, L^16) covers all 128 k's of row cl
        const float s_tot = (sp + __shfl_xor(sp, 16, 32)) * AINV;

        // logits: C layout — acc[n][v] is (M = v + hi*8, N = n*16 + cl)
        float p[8];
        #pragma unroll
        for (int v = 0; v < 8; ++v) p[v] = 0.0f;
        #pragma unroll
        for (int n = 0; n < 4; ++n) {
            const int col = n * 16 + cl;
            const float bb = b1s[col], ww = w2s[col];
            #pragma unroll
            for (int v = 0; v < 8; ++v) {
                const float x = acc[n][v] * AINV + bb;   // unscale before ReLU
                p[v] += (x > 0.0f ? x : 0.0f) * ww;
            }
        }
        #pragma unroll
        for (int m = 1; m <= 8; m <<= 1) {
            #pragma unroll
            for (int v = 0; v < 8; ++v) p[v] += __shfl_xor(p[v], m, 32);
        }

        // lane with cl<8 owns row M = hi*8 + cl
        float logit = p[0];
        #pragma unroll
        for (int v = 1; v < 8; ++v) logit = (cl == v) ? p[v] : logit;
        const int   M    = hi * 8 + cl;
        const float srow = __shfl(s_tot, M, 32);   // s_tot lives at lane==row for rows 0..15
        const int   j    = tile * 16 + M;

        float e = 0.0f;
        if (cl < 8 && j < HLEN && histRow[j] != tgt)
            e = __expf(logit);
        float es = e * srow;
        #pragma unroll
        for (int m = 1; m < 32; m <<= 1) {
            e  += __shfl_xor(e,  m, 32);
            es += __shfl_xor(es, m, 32);
        }
        if (lane == 0) {
            atomicAdd(&s_exp, e);     // ds_add_f32
            atomicAdd(&s_es,  es);
        }
    }

    __syncthreads();
    if (tid == 0) {
        const float pred = s_es / sqrtf(s_exp);          // beta = 0.5 normalization
        out[b] = 1.0f / (1.0f + __expf(-pred));
    }
}

extern "C" void kernel_launch(void* const* d_in, const int* in_sizes, int n_in,
                              void* d_out, int out_size, void* d_ws, size_t ws_size,
                              hipStream_t stream) {
    const int*   history     = (const int*)  d_in[0];
    const int*   target      = (const int*)  d_in[1];
    const int*   hist_region = (const int*)  d_in[2];
    const int*   tgt_region  = (const int*)  d_in[3];
    const float* W_hist      = (const float*)d_in[4];
    const float* W_tgt       = (const float*)d_in[5];
    const float* W_reg       = (const float*)d_in[6];
    const float* W1          = (const float*)d_in[7];
    const float* b1          = (const float*)d_in[8];
    const float* W2          = (const float*)d_in[9];
    float* out = (float*)d_out;

    const int B = in_sizes[1];   // 4096
    dim3 grid(B), block(64);
    nais_region_fused<<<grid, block, 0, stream>>>(
        history, target, hist_region, tgt_region,
        W_hist, W_tgt, W_reg, W1, b1, W2, out);
    (void)d_ws; (void)ws_size; (void)n_in; (void)out_size;
}
